// SCFFTBlock_82317343195166
// MI455X (gfx1250) — matrix operand order repaired and checked by reading, not compiled, not run on hardware
//
#include <hip/hip_runtime.h>
#include <hip/hip_bf16.h>
#include <math.h>
#include <stdint.h>

#define B_  16
#define T_  1024
#define D_  256
#define H_  2
#define DK_ 64
#define DV_ 64
#define C1_ 1024
#define C2_ 1024
#define K_  9
#define NEG_ (-1000000000.0f)
#define LN_EPS_ 1e-5f

typedef __attribute__((ext_vector_type(16))) __bf16 v16bf;
typedef __attribute__((ext_vector_type(8)))  __bf16 v8bf;
typedef __attribute__((ext_vector_type(8)))  float  v8f;

__device__ __forceinline__ float mishf(float x) {
    float sp = (x > 20.f) ? x : log1pf(__expf(x));
    return x * tanhf(sp);
}

// CDNA5 async copy: LDS[ldsOff] = MEM[g] (16 bytes, per-lane, tracked by ASYNCcnt)
__device__ __forceinline__ void async_ld16(uint32_t ldsOff, const void* g) {
    asm volatile("global_load_async_to_lds_b128 %0, %1, off"
                 :: "v"(ldsOff), "v"((unsigned long long)(uintptr_t)g)
                 : "memory");
}

// ---------------------------------------------------------------------------
// Batched bf16 GEMM, v_wmma_f32_16x16x32_bf16, f32 accumulate.
//   C[z] = epi( alpha * A[z] @ W[z]^T )   A: M x K (lda), W: N x K (ldw), bf16
// Batch offset: (z/zdiv)*S1 + (z%zdiv)*S2 (elements).
// Epilogue: v = alpha*acc*colScale[n] + bias[n] + res[m*ldres+n]; optional mish.
// OMODE: 0 = f32 store C[m*ldc+n], 1 = bf16 store C[m*ldc+n],
//        2 = bf16 transposed store C[n*ldc+m].
// Block: 128 threads = 4 waves; 64x64 tile; BK=32; LDS double buffered and
// filled with global_load_async_to_lds_b128 (next tile streams during wmma).
// All M,N multiples of 64 and K multiples of 32 here -> no bounds checks.
// ---------------------------------------------------------------------------
template<int OMODE>
__global__ __launch_bounds__(128)
void gemm_bf16(const __bf16* __restrict__ A, long long aS1, long long aS2, int lda,
               const __bf16* __restrict__ W, long long wS1, long long wS2, int ldw,
               void* __restrict__ Cv, long long cS1, long long cS2, int ldc,
               const float* __restrict__ bias, long long bS,
               const float* __restrict__ colScale, long long sS,
               const float* __restrict__ res, int ldres,
               int Kd, int zdiv, float alpha, int actMish)
{
    // row stride 40 elements (80 B): 16B-aligned chunks, conflict padding
    __shared__ __align__(16) __bf16 As[2][64][40];
    __shared__ __align__(16) __bf16 Bs[2][64][40];

    const int z = blockIdx.z;
    const long long zq = z / zdiv, zr = z % zdiv;
    A += zq * aS1 + zr * aS2;
    W += zq * wS1 + zr * wS2;
    char* Cb = (char*)Cv + (size_t)(zq * cS1 + zr * cS2) * (OMODE == 0 ? 4 : 2);
    if (bias)     bias     += (long long)z * bS;
    if (colScale) colScale += (long long)z * sS;

    const int m0 = blockIdx.y * 64, n0 = blockIdx.x * 64;
    const int tid  = threadIdx.x;
    const int wave = tid >> 5, lane = tid & 31;
    const int wm = (wave >> 1) * 32, wn = (wave & 1) * 32;
    const int half = lane >> 4, r = lane & 15;

    // this thread's two 16B chunks of a 64x32 tile: chunk q -> row q>>2, col (q&3)*8
    const int q0 = tid, q1 = tid + 128;
    const int ar0 = q0 >> 2, ac0 = (q0 & 3) * 8;
    const int ar1 = q1 >> 2, ac1 = (q1 & 3) * 8;

    v8f acc[2][2];
    const v8f vzero = {0.f,0.f,0.f,0.f,0.f,0.f,0.f,0.f};
    acc[0][0] = vzero; acc[0][1] = vzero; acc[1][0] = vzero; acc[1][1] = vzero;

    // stage tile k0 into LDS buffer `b` (4 async issues per thread)
    auto stage = [&](int k0, int b) {
        const uint32_t la = (uint32_t)(uintptr_t)&As[b][0][0];
        const uint32_t lb = (uint32_t)(uintptr_t)&Bs[b][0][0];
        async_ld16(la + (uint32_t)(ar0 * 80 + ac0 * 2), A + (long long)(m0 + ar0) * lda + k0 + ac0);
        async_ld16(la + (uint32_t)(ar1 * 80 + ac1 * 2), A + (long long)(m0 + ar1) * lda + k0 + ac1);
        async_ld16(lb + (uint32_t)(ar0 * 80 + ac0 * 2), W + (long long)(n0 + ar0) * ldw + k0 + ac0);
        async_ld16(lb + (uint32_t)(ar1 * 80 + ac1 * 2), W + (long long)(n0 + ar1) * ldw + k0 + ac1);
    };

    stage(0, 0);
    int bufi = 0;
    for (int k0 = 0; k0 < Kd; k0 += 32) {
        __syncthreads();                       // prev iteration's LDS reads done
        const bool nxt = (k0 + 32) < Kd;
        if (nxt) stage(k0 + 32, bufi ^ 1);     // stream next tile during compute
        if (nxt) asm volatile("s_wait_asynccnt 0x4" ::: "memory");
        else     asm volatile("s_wait_asynccnt 0x0" ::: "memory");
        __syncthreads();                       // everyone's current tile landed

        // A fragment (16-bit A 16x32): lane<16 -> K {0..7}U{16..23}; lane>=16 -> {8..15}U{24..31}
        v16bf af[2], bf[2];
        #pragma unroll
        for (int i = 0; i < 2; ++i) {
            const int row = wm + i * 16 + r;
            const v8bf c0 = *reinterpret_cast<const v8bf*>(&As[bufi][row][half * 8]);
            const v8bf c1 = *reinterpret_cast<const v8bf*>(&As[bufi][row][16 + half * 8]);
            #pragma unroll
            for (int e = 0; e < 8; ++e) { af[i][e] = c0[e]; af[i][e + 8] = c1[e]; }
        }
        // B fragment (16-bit B 32x16): lane<16 -> K {0..7}U{16..23}; lane>=16 -> {8..15}U{24..31}
        #pragma unroll
        for (int j = 0; j < 2; ++j) {
            const int col = wn + j * 16 + r;
            const v8bf c0 = *reinterpret_cast<const v8bf*>(&Bs[bufi][col][half * 8]);
            const v8bf c1 = *reinterpret_cast<const v8bf*>(&Bs[bufi][col][16 + half * 8]);
            #pragma unroll
            for (int e = 0; e < 8; ++e) { bf[j][e] = c0[e]; bf[j][e + 8] = c1[e]; }
        }

        #pragma unroll
        for (int i = 0; i < 2; ++i)
            #pragma unroll
            for (int j = 0; j < 2; ++j)
                acc[i][j] = __builtin_amdgcn_wmma_f32_16x16x32_bf16(
                    false, af[i], false, bf[j], (short)0, acc[i][j], false, false);
        bufi ^= 1;
    }

    // epilogue: D row = e + 8*half, col = lane&15
    #pragma unroll
    for (int i = 0; i < 2; ++i) {
        #pragma unroll
        for (int j = 0; j < 2; ++j) {
            const int colg = n0 + wn + j * 16 + r;
            const float cs  = colScale ? colScale[colg] : 1.f;
            const float bvv = bias ? bias[colg] : 0.f;
            #pragma unroll
            for (int e = 0; e < 8; ++e) {
                const int rowg = m0 + wm + i * 16 + half * 8 + e;
                float v = acc[i][j][e] * alpha * cs + bvv;
                if (res) v += res[(long long)rowg * ldres + colg];
                if (actMish) v = mishf(v);
                if (OMODE == 0)      ((float*)Cb)[(long long)rowg * ldc + colg]  = v;
                else if (OMODE == 1) ((__bf16*)Cb)[(long long)rowg * ldc + colg] = (__bf16)v;
                else                 ((__bf16*)Cb)[(long long)colg * ldc + rowg] = (__bf16)v;
            }
        }
    }
}

// ---------------------------------------------------------------------------
// elementwise f32 -> bf16
__global__ __launch_bounds__(256)
void cvt_kernel(const float* __restrict__ in, __bf16* __restrict__ out, long long n)
{
    long long i = (long long)blockIdx.x * 256 + threadIdx.x;
    const long long stride = (long long)gridDim.x * 256;
    for (; i < n; i += stride) out[i] = (__bf16)in[i];
}

// p_w (B,C2,C1) -> pwT (B,C1,C2) bf16 (tiled transpose)
__global__ __launch_bounds__(256)
void transpose_cvt_kernel(const float* __restrict__ src, __bf16* __restrict__ dst)
{
    __shared__ float tile[32][33];
    const int b = blockIdx.z;
    const int c0 = blockIdx.x * 32;   // C1 block
    const int r0 = blockIdx.y * 32;   // C2 block
    for (int i = threadIdx.y; i < 32; i += 8)
        tile[i][threadIdx.x] = src[(long long)b * C2_ * C1_ + (long long)(r0 + i) * C1_ + c0 + threadIdx.x];
    __syncthreads();
    for (int i = threadIdx.y; i < 32; i += 8)
        dst[(long long)b * C1_ * C2_ + (long long)(c0 + i) * C2_ + r0 + threadIdx.x] =
            (__bf16)tile[threadIdx.x][i];
}

// ---------------------------------------------------------------------------
// Row softmax over T=1024 with attention mask; writes f32 (output) + bf16 (GEMM A)
__global__ __launch_bounds__(256)
void softmax_kernel(float* __restrict__ attn, __bf16* __restrict__ attnb,
                    const unsigned char* __restrict__ smask)
{
    __shared__ float sm[256];
    const long long row = blockIdx.x;                 // b*H*T + h*T + q
    const int b = (int)(row >> 11);
    const int q = (int)(row & (T_ - 1));
    float*  p  = attn  + row * (long long)T_;
    __bf16* pb = attnb + row * (long long)T_;
    const unsigned char* mrow = smask + ((long long)b * T_ + q) * (long long)T_;

    const int t = threadIdx.x;
    float v[4];
    float mx = -1e30f;
    #pragma unroll
    for (int u = 0; u < 4; ++u) {
        const int c = t * 4 + u;
        float s = p[c];
        if (mrow[c]) s = NEG_;
        v[u] = s;
        mx = fmaxf(mx, s);
    }
    sm[t] = mx; __syncthreads();
    for (int s2 = 128; s2 > 0; s2 >>= 1) { if (t < s2) sm[t] = fmaxf(sm[t], sm[t + s2]); __syncthreads(); }
    mx = sm[0]; __syncthreads();

    float sum = 0.f;
    #pragma unroll
    for (int u = 0; u < 4; ++u) { v[u] = __expf(v[u] - mx); sum += v[u]; }
    sm[t] = sum; __syncthreads();
    for (int s2 = 128; s2 > 0; s2 >>= 1) { if (t < s2) sm[t] += sm[t + s2]; __syncthreads(); }
    const float inv = 1.f / sm[0];

    #pragma unroll
    for (int u = 0; u < 4; ++u) {
        const float o = v[u] * inv;
        p[t * 4 + u]  = o;
        pb[t * 4 + u] = (__bf16)o;
    }
}

// ---------------------------------------------------------------------------
// LayerNorm over D=256 + token mask; optional extra bf16 copy for GEMM A use.
__global__ __launch_bounds__(256)
void ln_kernel(const float* __restrict__ in, float* __restrict__ out,
               __bf16* __restrict__ outb,
               const float* __restrict__ g, const float* __restrict__ bt,
               const unsigned char* __restrict__ mask)
{
    __shared__ float sm[256];
    const long long row = blockIdx.x;
    const int t = threadIdx.x;
    const float xv = in[row * D_ + t];

    sm[t] = xv; __syncthreads();
    for (int s2 = 128; s2 > 0; s2 >>= 1) { if (t < s2) sm[t] += sm[t + s2]; __syncthreads(); }
    const float mean = sm[0] * (1.f / D_); __syncthreads();

    const float d = xv - mean;
    sm[t] = d * d; __syncthreads();
    for (int s2 = 128; s2 > 0; s2 >>= 1) { if (t < s2) sm[t] += sm[t + s2]; __syncthreads(); }
    const float var = sm[0] * (1.f / D_);

    float y = d * rsqrtf(var + LN_EPS_) * g[t] + bt[t];
    if (mask[row]) y = 0.f;
    out[row * D_ + t] = y;
    if (outb) outb[row * D_ + t] = (__bf16)y;
}

// ---------------------------------------------------------------------------
// l2 norms. norm1: over C1 per (b,k) for d_w. norm2: over C2 per (b,c1) for p_w.
__global__ void norm1_kernel(const float* __restrict__ d_w, float* __restrict__ invn1)
{
    const int id = threadIdx.x;
    if (id >= B_ * K_) return;
    const int b = id / K_, k = id % K_;
    float s = 0.f;
    for (int c = 0; c < C1_; ++c) {
        const float w = d_w[((long long)b * C1_ + c) * K_ + k];
        s += w * w;
    }
    invn1[id] = 1.f / fmaxf(sqrtf(s), 1e-12f);
}

__global__ __launch_bounds__(256)
void norm2_kernel(const float* __restrict__ p_w, float* __restrict__ invn2)
{
    const int id = blockIdx.x * 256 + threadIdx.x;
    const int b = id / C1_, c1 = id % C1_;
    float s = 0.f;
    for (int c2 = 0; c2 < C2_; ++c2) {
        const float w = p_w[((long long)b * C2_ + c2) * C1_ + c1];
        s += w * w;
    }
    invn2[id] = 1.f / fmaxf(sqrtf(s), 1e-12f);
}

// ---------------------------------------------------------------------------
// Depthwise conv over time (K=9, pad 4), normalized weights on the fly;
// p_g folded in (it scales the pointwise contraction axis). bf16 in/out.
__global__ __launch_bounds__(256)
void dwconv_kernel(const __bf16* __restrict__ h, const float* __restrict__ d_w,
                   const float* __restrict__ d_g, const float* __restrict__ d_b,
                   const float* __restrict__ invn1, const float* __restrict__ p_g,
                   __bf16* __restrict__ dep)
{
    const int t = blockIdx.x, b = blockIdx.y;
    for (int ps = 0; ps < 4; ++ps) {
        const int c = threadIdx.x + ps * 256;
        const float gc = d_g[b * C1_ + c];
        float acc = 0.f;
        #pragma unroll
        for (int k = 0; k < K_; ++k) {
            const int tt = t + k - 4;
            if (tt >= 0 && tt < T_) {
                const float wv = d_w[((long long)b * C1_ + c) * K_ + k] * invn1[b * K_ + k] * gc;
                acc += (float)h[((long long)(b * T_ + tt)) * C1_ + c] * wv;
            }
        }
        acc += d_b[b * C1_ + c];
        dep[((long long)(b * T_ + t)) * C1_ + c] = (__bf16)(acc * p_g[b * C2_ + c]);
    }
}

// ---------------------------------------------------------------------------
extern "C" void kernel_launch(void* const* d_in, const int* in_sizes, int n_in,
                              void* d_out, int out_size, void* d_ws, size_t ws_size,
                              hipStream_t stream)
{
    const float* x    = (const float*)d_in[0];
    const float* d_w  = (const float*)d_in[1];
    const float* d_g  = (const float*)d_in[2];
    const float* d_b  = (const float*)d_in[3];
    const float* p_w  = (const float*)d_in[4];
    const float* p_g  = (const float*)d_in[5];
    const float* p_b  = (const float*)d_in[6];
    const unsigned char* mask  = (const unsigned char*)d_in[7];
    const unsigned char* smask = (const unsigned char*)d_in[8];
    const float* Wq = (const float*)d_in[9];   const float* bq  = (const float*)d_in[10];
    const float* Wk = (const float*)d_in[11];  const float* bk  = (const float*)d_in[12];
    const float* Wv = (const float*)d_in[13];  const float* bv  = (const float*)d_in[14];
    const float* Wo = (const float*)d_in[15];  const float* bo  = (const float*)d_in[16];
    const float* g0 = (const float*)d_in[17];  const float* b0  = (const float*)d_in[18];
    const float* w1w = (const float*)d_in[19]; const float* w1b = (const float*)d_in[20];
    const float* w2w = (const float*)d_in[21]; const float* w2b = (const float*)d_in[22];
    const float* g1 = (const float*)d_in[23];  const float* b1  = (const float*)d_in[24];

    float* out  = (float*)d_out;
    float* attn = out + (size_t)B_ * T_ * D_;

    const size_t MT = (size_t)B_ * T_;               // 16384
    char* wsb = (char*)d_ws;
    auto alloc = [&](size_t bytes) -> void* {
        void* p = wsb; wsb += (bytes + 255) & ~(size_t)255; return p;
    };
    __bf16* xb    = (__bf16*)alloc(MT * D_ * 2);
    __bf16* wqb   = (__bf16*)alloc((size_t)128 * D_ * 2);
    __bf16* wkb   = (__bf16*)alloc((size_t)128 * D_ * 2);
    __bf16* wvb   = (__bf16*)alloc((size_t)128 * D_ * 2);
    __bf16* wob   = (__bf16*)alloc((size_t)D_ * 128 * 2);
    __bf16* w1wb  = (__bf16*)alloc((size_t)C1_ * D_ * 2);
    __bf16* w2wb  = (__bf16*)alloc((size_t)D_ * C2_ * 2);
    __bf16* pwT   = (__bf16*)alloc((size_t)B_ * C1_ * C2_ * 2);
    __bf16* qb    = (__bf16*)alloc(MT * 128 * 2);
    __bf16* kb    = (__bf16*)alloc(MT * 128 * 2);
    __bf16* vT    = (__bf16*)alloc((size_t)128 * MT * 2);
    __bf16* attnb = (__bf16*)alloc((size_t)B_ * H_ * T_ * T_ * 2);
    __bf16* ctxb  = (__bf16*)alloc(MT * 128 * 2);
    float*  apre  = (float*)alloc(MT * D_ * 4);
    float*  a     = (float*)alloc(MT * D_ * 4);
    __bf16* ab    = (__bf16*)alloc(MT * D_ * 2);
    __bf16* hb    = (__bf16*)alloc(MT * C1_ * 2);
    __bf16* depb  = (__bf16*)alloc(MT * C1_ * 2);
    __bf16* ptb   = (__bf16*)alloc(MT * C2_ * 2);
    float*  o2    = (float*)alloc(MT * D_ * 4);
    float*  invn1 = (float*)alloc(256 * 4);
    float*  invn2 = (float*)alloc((size_t)B_ * C1_ * 4);

    auto cvt = [&](const float* in, __bf16* o, long long n) {
        int grid = (int)((n + 1023) / 1024); if (grid > 4096) grid = 4096; if (grid < 1) grid = 1;
        cvt_kernel<<<dim3(grid), dim3(256), 0, stream>>>(in, o, n);
    };

    // one-time operand conversions
    cvt(x, xb, (long long)MT * D_);
    cvt(Wq, wqb, 128 * D_);
    cvt(Wk, wkb, 128 * D_);
    cvt(Wv, wvb, 128 * D_);
    cvt(Wo, wob, D_ * 128);
    cvt(w1w, w1wb, C1_ * D_);
    cvt(w2w, w2wb, D_ * C2_);
    transpose_cvt_kernel<<<dim3(32, 32, B_), dim3(32, 8), 0, stream>>>(p_w, pwT);

    // weight-norm reductions
    norm1_kernel<<<dim3(1), dim3(256), 0, stream>>>(d_w, invn1);
    norm2_kernel<<<dim3(64), dim3(256), 0, stream>>>(p_w, invn2);

    const dim3 blk(128);
    const long long TT = (long long)T_ * T_;

    // Q,K projections -> bf16 (b,t,hd); V -> bf16 transposed vT[hd][b*T+t]
    gemm_bf16<1><<<dim3(2, 256, 1), blk, 0, stream>>>(
        xb, 0, 0, D_,  wqb, 0, 0, D_,  qb, 0, 0, 128,
        bq, 0, nullptr, 0, nullptr, 0, D_, 1, 1.f, 0);
    gemm_bf16<1><<<dim3(2, 256, 1), blk, 0, stream>>>(
        xb, 0, 0, D_,  wkb, 0, 0, D_,  kb, 0, 0, 128,
        bk, 0, nullptr, 0, nullptr, 0, D_, 1, 1.f, 0);
    gemm_bf16<2><<<dim3(2, 256, 1), blk, 0, stream>>>(
        xb, 0, 0, D_,  wvb, 0, 0, D_,  vT, 0, 0, (int)MT,
        bv, 0, nullptr, 0, nullptr, 0, D_, 1, 1.f, 0);

    // scores[b,h] = (q @ k^T)/8 -> f32 straight into attn output region
    gemm_bf16<0><<<dim3(16, 16, B_ * H_), blk, 0, stream>>>(
        qb, (long long)T_ * 128, 64, 128,
        kb, (long long)T_ * 128, 64, 128,
        attn, 2 * TT, TT, T_,
        nullptr, 0, nullptr, 0, nullptr, 0,
        DK_, H_, 0.125f, 0);

    // softmax (+ slf_attn_mask) -> final attn (f32) + attnb (bf16)
    softmax_kernel<<<dim3(B_ * H_ * T_), dim3(256), 0, stream>>>(attn, attnb, smask);

    // ctx[b,h] = attn @ v   via vT as N x K:  W[d][t] = vT[(h*64+d)][b*T + t]
    gemm_bf16<1><<<dim3(1, 16, B_ * H_), blk, 0, stream>>>(
        attnb, 2 * TT, TT, T_,
        vT, 1024, (long long)64 * MT, (int)MT,
        ctxb, (long long)T_ * 128, 64, 128,
        nullptr, 0, nullptr, 0, nullptr, 0,
        T_, H_, 1.f, 0);

    // a_pre = ctx @ Wo^T + bo + x ; LN(g0,b0)+mask -> a (f32) and ab (bf16)
    gemm_bf16<0><<<dim3(4, 256, 1), blk, 0, stream>>>(
        ctxb, 0, 0, 128,  wob, 0, 0, 128,  apre, 0, 0, D_,
        bo, 0, nullptr, 0, x, D_, 128, 1, 1.f, 0);
    ln_kernel<<<dim3(B_ * T_), dim3(256), 0, stream>>>(apre, a, ab, g0, b0, mask);

    // h = mish(a @ w1^T + w1_b) -> bf16 (b,t,c)
    gemm_bf16<1><<<dim3(16, 256, 1), blk, 0, stream>>>(
        ab, 0, 0, D_,  w1wb, 0, 0, D_,  hb, 0, 0, C1_,
        w1b, 0, nullptr, 0, nullptr, 0, D_, 1, 1.f, 1);

    // depthwise conv (+ d_b, * p_g fold) -> bf16
    dwconv_kernel<<<dim3(T_, B_), dim3(256), 0, stream>>>(
        hb, d_w, d_g, d_b, invn1, p_g, depb);

    // pointwise: pt[b,t,o] = (sum_c dep'[b,t,c]*p_w[b,c,o])*invn2[b,o] + p_b[b,o]
    gemm_bf16<1><<<dim3(16, 16, B_), blk, 0, stream>>>(
        depb, (long long)T_ * C1_, 0, C1_,
        pwT, (long long)C1_ * C2_, 0, C2_,
        ptb, (long long)T_ * C2_, 0, C2_,
        p_b, C2_, invn2, C1_, nullptr, 0,
        C1_, 1, 1.f, 0);

    // o2 = pt @ w2^T + w2_b + a ; LN(g1,b1)+mask -> out
    gemm_bf16<0><<<dim3(4, 256, 1), blk, 0, stream>>>(
        ptb, 0, 0, C2_,  w2wb, 0, 0, C2_,  o2, 0, 0, D_,
        w2b, 0, nullptr, 0, a, D_, C2_, 1, 1.f, 0);
    ln_kernel<<<dim3(B_ * T_), dim3(256), 0, stream>>>(o2, out, nullptr, g1, b1, mask);
}
